// DGCNN_23691039605450
// MI455X (gfx1250) — matrix operand order repaired and checked by reading, not compiled, not run on hardware
//
#include <hip/hip_runtime.h>

// DGCNN inference for MI455X (gfx1250, wave32).
// EdgeConv MLP rewritten as two dense WMMA GEMMs (U = X*(Wtop-Wbot), V = X*Wbot)
// + gather-max. All GEMM operands pre-packed to f16 fragment-friendly layouts so
// the WMMA inner loop is pure b128 loads + v_wmma_f32_16x16x32_f16.

#define NPTS  32768   // B*P
#define PPC   4096    // points per cloud
#define NKNN  20
#define HF    64
#define NCLS  40
#define NEG   0.2f

typedef __attribute__((ext_vector_type(16))) _Float16 v16h;
typedef __attribute__((ext_vector_type(8)))  _Float16 v8h;
typedef __attribute__((ext_vector_type(8)))  float    v8f;

__device__ __forceinline__ float lrelu(float v) { return v > 0.f ? v : NEG * v; }

// --------------------------------------------------------------------------
// kNN: per-thread query, candidates tiled through LDS (128/tile) with
// precomputed norms; exact fp32 distances; register-resident sorted top-20.
// --------------------------------------------------------------------------
template<int D>
__global__ __launch_bounds__(256) void knn_kernel(const float* __restrict__ X, int ldx,
                                                  int* __restrict__ nbr) {
  constexpr int TJ = 128;
  __shared__ float s_c[TJ * D];
  __shared__ float s_n[TJ];
  const int tid   = threadIdx.x;
  const int cloud = blockIdx.x >> 4;                      // 16 blocks per cloud
  const int qg    = cloud * PPC + ((blockIdx.x & 15) * 256 + tid);

  float q[D]; float qn = 0.f;
  #pragma unroll
  for (int m = 0; m < D; ++m) { q[m] = X[(size_t)qg * ldx + m]; qn += q[m] * q[m]; }

  float bd[NKNN]; int bj[NKNN];
  #pragma unroll
  for (int t = 0; t < NKNN; ++t) { bd[t] = 3.0e38f; bj[t] = qg; }

  for (int j0 = 0; j0 < PPC; j0 += TJ) {
    __syncthreads();
    for (int t = tid; t < TJ * D; t += 256) {
      int r = t / D;
      s_c[t] = X[(size_t)(cloud * PPC + j0 + r) * ldx + (t - r * D)];
    }
    __syncthreads();
    if (tid < TJ) {
      float n = 0.f;
      #pragma unroll
      for (int m = 0; m < D; ++m) { float v = s_c[tid * D + m]; n += v * v; }
      s_n[tid] = n;
    }
    __syncthreads();
    for (int r = 0; r < TJ; ++r) {
      float dot = 0.f;
      #pragma unroll
      for (int m = 0; m < D; ++m) dot += q[m] * s_c[r * D + m];
      float dist = qn + s_n[r] - 2.f * dot;
      int jg = cloud * PPC + j0 + r;
      if (jg != qg && dist < bd[NKNN - 1]) {
        bd[NKNN - 1] = dist; bj[NKNN - 1] = jg;
        #pragma unroll
        for (int t = NKNN - 1; t > 0; --t) {       // bubble new element into place
          if (bd[t] < bd[t - 1]) {
            float td = bd[t]; bd[t] = bd[t - 1]; bd[t - 1] = td;
            int   tj = bj[t]; bj[t] = bj[t - 1]; bj[t - 1] = tj;
          }
        }
      }
    }
  }
  #pragma unroll
  for (int t = 0; t < NKNN; ++t) nbr[(size_t)qg * NKNN + t] = bj[t];
}

// --------------------------------------------------------------------------
// Packing kernels (one-shot, bandwidth-bound).
// --------------------------------------------------------------------------
// f32 [M x D] (row stride lda) -> f16 [M x Kp], zero-padded columns D..Kp-1.
__global__ void pack_a_kernel(const float* __restrict__ src, int lda, int D,
                              _Float16* __restrict__ dst, int Kp, int total) {
  int t = blockIdx.x * 256 + threadIdx.x;
  if (t >= total) return;
  int row = t / Kp, k = t - row * Kp;
  dst[t] = (k < D) ? (_Float16)src[(size_t)row * lda + k] : (_Float16)0.f;
}

// W [Kin x Nout] f32 -> Wt [Noutp x Kp] f16, transposed, zero-padded.
__global__ void pack_wt_kernel(const float* __restrict__ W, int Kin, int Nout,
                               _Float16* __restrict__ Wt, int Kp, int total) {
  int t = blockIdx.x * 256 + threadIdx.x;
  if (t >= total) return;
  int n = t / Kp, k = t - n * Kp;
  Wt[t] = (n < Nout && k < Kin) ? (_Float16)W[(size_t)k * Nout + n] : (_Float16)0.f;
}

// Edge MLP weights W [2d x H]: Wut = (Wtop - Wbot)^T, Wvt = Wbot^T, f16 [H x Dp].
__global__ void pack_wedge_kernel(const float* __restrict__ W, int d,
                                  _Float16* __restrict__ Wut, _Float16* __restrict__ Wvt,
                                  int Dp, int total) {
  int t = blockIdx.x * 256 + threadIdx.x;
  if (t >= total) return;
  int n = t / Dp, k = t - n * Dp;
  float top = (k < d) ? W[(size_t)k * HF + n] : 0.f;
  float bot = (k < d) ? W[(size_t)(k + d) * HF + n] : 0.f;
  Wut[t] = (_Float16)(top - bot);
  Wvt[t] = (_Float16)bot;
}

// --------------------------------------------------------------------------
// Dense GEMM: out[M x Nout] = act(A16[M x Kp] * Wt[Noutp x Kp]^T + bias)
// A16 row-major f16 (lda = Kp multiple of 32, 16B aligned);
// Wt transposed f16 (ldw = Kp, Noutp rows padded to multiple of 64, zero-filled).
// One wave per 16x16 output tile; 4 waves/block. Inner loop: 4x b128 + 1 WMMA.
// --------------------------------------------------------------------------
__global__ __launch_bounds__(128) void gemm_wmma(const _Float16* __restrict__ A16, int lda,
                                                 const _Float16* __restrict__ Wt, int ldw,
                                                 const float* __restrict__ bias,
                                                 void* __restrict__ out, int ldo,
                                                 int Kp, int Nout, int act, int out_f16) {
  const int lane  = threadIdx.x & 31;
  const int wave  = threadIdx.x >> 5;
  const int lrow  = lane & 15;
  const int lhalf = lane >> 4;
  const int m0    = blockIdx.x * 16;
  const int n0    = (blockIdx.y * 4 + wave) * 16;
  const int col   = n0 + lrow;

  // A 16x32 f16 fragment: lane half h needs k runs [k0+8h, +8) and [k0+16+8h, +8)
  const _Float16* ap = A16 + (size_t)(m0 + lrow) * lda + lhalf * 8;
  // B 32x16 f16 fragment: lane half h needs k runs [k0+8h, +8) and [k0+16+8h, +8)
  const _Float16* bp = Wt + (size_t)col * ldw + lhalf * 8;

  v8f c = {};
  for (int k0 = 0; k0 < Kp; k0 += 32) {
    v8h alo = *(const v8h*)(ap);
    v8h ahi = *(const v8h*)(ap + 16);
    v8h blo = *(const v8h*)(bp);
    v8h bhi = *(const v8h*)(bp + 16);
    ap += 32; bp += 32;
    v16h a = __builtin_shufflevector(alo, ahi, 0, 1, 2, 3, 4, 5, 6, 7,
                                     8, 9, 10, 11, 12, 13, 14, 15);
    v16h b = __builtin_shufflevector(blo, bhi, 0, 1, 2, 3, 4, 5, 6, 7,
                                     8, 9, 10, 11, 12, 13, 14, 15);
    c = __builtin_amdgcn_wmma_f32_16x16x32_f16(false, a, false, b, (short)0, c,
                                               false, false);
  }
  if (col < Nout) {
    float bv = bias ? bias[col] : 0.f;
    #pragma unroll
    for (int p = 0; p < 8; ++p) {        // C/D: VGPR p -> row m0 + lhalf*8 + p
      int row = m0 + lhalf * 8 + p;
      float v = c[p] + bv;
      if (act) v = lrelu(v);
      if (out_f16) ((_Float16*)out)[(size_t)row * ldo + col] = (_Float16)v;
      else         ((float*)out)[(size_t)row * ldo + col] = v;
    }
  }
}

// --------------------------------------------------------------------------
// h[i][c] = max_{j in nbr(i)} LReLU(U[i][c] + V[j][c] + bias[c]); one wave/point.
// --------------------------------------------------------------------------
__global__ __launch_bounds__(256) void edge_max_kernel(const float* __restrict__ U,
                                                       const float* __restrict__ V,
                                                       const int* __restrict__ nbr,
                                                       const float* __restrict__ bias,
                                                       float* __restrict__ out,
                                                       int ldo, int col_off) {
  const int wave = threadIdx.x >> 5, lane = threadIdx.x & 31;
  const int i = blockIdx.x * 8 + wave;
  const float u0 = U[(size_t)i * HF + lane];
  const float u1 = U[(size_t)i * HF + 32 + lane];
  const float b0 = bias[lane], b1 = bias[lane + 32];
  float m0 = -3.0e38f, m1 = -3.0e38f;
  const int* nb = nbr + (size_t)i * NKNN;
  #pragma unroll 4
  for (int k = 0; k < NKNN; ++k) {
    int j = nb[k];
    float e0 = lrelu(u0 + V[(size_t)j * HF + lane] + b0);
    float e1 = lrelu(u1 + V[(size_t)j * HF + 32 + lane] + b1);
    m0 = fmaxf(m0, e0); m1 = fmaxf(m1, e1);
  }
  out[(size_t)i * ldo + col_off + lane] = m0;
  out[(size_t)i * ldo + col_off + lane + 32] = m1;
}

// --------------------------------------------------------------------------
// In-place log_softmax over NCLS=40 values; one wave per point.
// --------------------------------------------------------------------------
__global__ __launch_bounds__(256) void logsoftmax_kernel(float* __restrict__ out, int C) {
  const int wave = threadIdx.x >> 5, lane = threadIdx.x & 31;
  const int i = blockIdx.x * 8 + wave;
  float* row = out + (size_t)i * C;
  float v0 = (lane < C) ? row[lane] : -3.0e38f;
  float v1 = (lane + 32 < C) ? row[lane + 32] : -3.0e38f;
  float m = fmaxf(v0, v1);
  #pragma unroll
  for (int o = 16; o > 0; o >>= 1) m = fmaxf(m, __shfl_xor(m, o, 32));
  float s = ((lane < C) ? __expf(v0 - m) : 0.f) +
            ((lane + 32 < C) ? __expf(v1 - m) : 0.f);
  #pragma unroll
  for (int o = 16; o > 0; o >>= 1) s += __shfl_xor(s, o, 32);
  float lse = m + __logf(s);
  if (lane < C)      row[lane]      = v0 - lse;
  if (lane + 32 < C) row[lane + 32] = v1 - lse;
}

// --------------------------------------------------------------------------
extern "C" void kernel_launch(void* const* d_in, const int* in_sizes, int n_in,
                              void* d_out, int out_size, void* d_ws, size_t ws_size,
                              hipStream_t stream) {
  (void)in_sizes; (void)n_in; (void)out_size; (void)ws_size;
  const float* x   = (const float*)d_in[0];
  // d_in[1] = batch (implied by fixed layout, unused)
  const float* W1  = (const float*)d_in[2];  const float* b1  = (const float*)d_in[3];
  const float* W2  = (const float*)d_in[4];  const float* b2  = (const float*)d_in[5];
  const float* W3  = (const float*)d_in[6];  const float* b3  = (const float*)d_in[7];
  const float* Wl  = (const float*)d_in[8];  const float* bl  = (const float*)d_in[9];
  const float* Wm1 = (const float*)d_in[10]; const float* bm1 = (const float*)d_in[11];
  const float* Wm2 = (const float*)d_in[12]; const float* bm2 = (const float*)d_in[13];
  const float* Wo  = (const float*)d_in[14]; const float* bo  = (const float*)d_in[15];
  float* out = (float*)d_out;

  // Workspace (~150 MB)
  char* ws = (char*)d_ws;
  size_t off = 0;
  auto alloc = [&](size_t bytes) -> void* {
    void* p = ws + off; off += (bytes + 255) & ~(size_t)255; return p;
  };
  int*      idx   = (int*)      alloc((size_t)NPTS * NKNN * 4);
  float*    Hcat  = (float*)    alloc((size_t)NPTS * 192 * 4);
  float*    U     = (float*)    alloc((size_t)NPTS * HF * 4);
  float*    V     = (float*)    alloc((size_t)NPTS * HF * 4);
  _Float16* X16   = (_Float16*) alloc((size_t)NPTS * 192 * 2);   // reused per pack
  _Float16* F1h   = (_Float16*) alloc((size_t)NPTS * 1024 * 2);
  _Float16* F2h   = (_Float16*) alloc((size_t)NPTS * 256 * 2);
  _Float16* F3h   = (_Float16*) alloc((size_t)NPTS * 128 * 2);
  _Float16* Wut1  = (_Float16*) alloc(64 * 32 * 2);
  _Float16* Wvt1  = (_Float16*) alloc(64 * 32 * 2);
  _Float16* Wut2  = (_Float16*) alloc(64 * 64 * 2);
  _Float16* Wvt2  = (_Float16*) alloc(64 * 64 * 2);
  _Float16* Wut3  = (_Float16*) alloc(64 * 64 * 2);
  _Float16* Wvt3  = (_Float16*) alloc(64 * 64 * 2);
  _Float16* Wlt   = (_Float16*) alloc((size_t)1024 * 192 * 2);
  _Float16* Wm1t  = (_Float16*) alloc((size_t)256 * 1024 * 2);
  _Float16* Wm2t  = (_Float16*) alloc((size_t)128 * 256 * 2);
  _Float16* Wot   = (_Float16*) alloc((size_t)64 * 128 * 2);     // 40 -> 64 padded rows

  const int MT = NPTS / 16;   // 2048 M-tiles
  auto blks = [](int total) { return (total + 255) / 256; };

  // ---- weight packs (tiny, once per call) ----
  pack_wedge_kernel<<<blks(64 * 32), 256, 0, stream>>>(W1, 3, Wut1, Wvt1, 32, 64 * 32);
  pack_wedge_kernel<<<blks(64 * 64), 256, 0, stream>>>(W2, 64, Wut2, Wvt2, 64, 64 * 64);
  pack_wedge_kernel<<<blks(64 * 64), 256, 0, stream>>>(W3, 64, Wut3, Wvt3, 64, 64 * 64);
  pack_wt_kernel<<<blks(1024 * 192), 256, 0, stream>>>(Wl, 192, 1024, Wlt, 192, 1024 * 192);
  pack_wt_kernel<<<blks(256 * 1024), 256, 0, stream>>>(Wm1, 1024, 256, Wm1t, 1024, 256 * 1024);
  pack_wt_kernel<<<blks(128 * 256), 256, 0, stream>>>(Wm2, 256, 128, Wm2t, 256, 128 * 256);
  pack_wt_kernel<<<blks(64 * 128), 256, 0, stream>>>(Wo, 128, NCLS, Wot, 128, 64 * 128);

  // ---- EdgeConv layer 1 (features = xyz, d=3 -> Kp=32) -> Hcat[:, 0:64] ----
  knn_kernel<3><<<128, 256, 0, stream>>>(x, 3, idx);
  pack_a_kernel<<<blks(NPTS * 32), 256, 0, stream>>>(x, 3, 3, X16, 32, NPTS * 32);
  gemm_wmma<<<dim3(MT, 1), 128, 0, stream>>>(X16, 32, Wut1, 32, nullptr, U, HF, 32, HF, 0, 0);
  gemm_wmma<<<dim3(MT, 1), 128, 0, stream>>>(X16, 32, Wvt1, 32, nullptr, V, HF, 32, HF, 0, 0);
  edge_max_kernel<<<NPTS / 8, 256, 0, stream>>>(U, V, idx, b1, Hcat, 192, 0);

  // ---- EdgeConv layer 2 (features = Hcat[:,0:64]) -> Hcat[:,64:128] ----
  knn_kernel<64><<<128, 256, 0, stream>>>(Hcat, 192, idx);
  pack_a_kernel<<<blks(NPTS * 64), 256, 0, stream>>>(Hcat, 192, 64, X16, 64, NPTS * 64);
  gemm_wmma<<<dim3(MT, 1), 128, 0, stream>>>(X16, 64, Wut2, 64, nullptr, U, HF, 64, HF, 0, 0);
  gemm_wmma<<<dim3(MT, 1), 128, 0, stream>>>(X16, 64, Wvt2, 64, nullptr, V, HF, 64, HF, 0, 0);
  edge_max_kernel<<<NPTS / 8, 256, 0, stream>>>(U, V, idx, b2, Hcat, 192, 64);

  // ---- EdgeConv layer 3 (features = Hcat[:,64:128]) -> Hcat[:,128:192] ----
  knn_kernel<64><<<128, 256, 0, stream>>>(Hcat + 64, 192, idx);
  pack_a_kernel<<<blks(NPTS * 64), 256, 0, stream>>>(Hcat + 64, 192, 64, X16, 64, NPTS * 64);
  gemm_wmma<<<dim3(MT, 1), 128, 0, stream>>>(X16, 64, Wut3, 64, nullptr, U, HF, 64, HF, 0, 0);
  gemm_wmma<<<dim3(MT, 1), 128, 0, stream>>>(X16, 64, Wvt3, 64, nullptr, V, HF, 64, HF, 0, 0);
  edge_max_kernel<<<NPTS / 8, 256, 0, stream>>>(U, V, idx, b3, Hcat, 192, 128);

  // ---- Head MLPs (intermediates stay f16) ----
  pack_a_kernel<<<blks(NPTS * 192), 256, 0, stream>>>(Hcat, 192, 192, X16, 192, NPTS * 192);
  gemm_wmma<<<dim3(MT, 16), 128, 0, stream>>>(X16, 192, Wlt, 192, bl, F1h, 1024, 192, 1024, 1, 1);
  gemm_wmma<<<dim3(MT, 4), 128, 0, stream>>>(F1h, 1024, Wm1t, 1024, bm1, F2h, 256, 1024, 256, 1, 1);
  gemm_wmma<<<dim3(MT, 2), 128, 0, stream>>>(F2h, 256, Wm2t, 256, bm2, F3h, 128, 256, 128, 1, 1);
  gemm_wmma<<<dim3(MT, 1), 128, 0, stream>>>(F3h, 128, Wot, 128, bo, out, NCLS, 128, NCLS, 0, 0);
  logsoftmax_kernel<<<NPTS / 8, 256, 0, stream>>>(out, NCLS);
}
